// Mtcnn_19739669692471
// MI455X (gfx1250) — matrix operand order repaired and checked by reading, not compiled, not run on hardware
//
#include <hip/hip_runtime.h>
#include <hip/hip_bf16.h>
#include <math.h>

typedef __attribute__((ext_vector_type(16))) _Float16 v16h;
typedef __attribute__((ext_vector_type(8)))  float    v8f;

#define MAX_KP 1152   // max CIN*KH*KW in this net (ONet fc: 1152, already mult of 64)

// ---------------------------------------------------------------------------
// Implicit-GEMM conv / dense on the WMMA pipe (wave32, 16x16x32 f16 -> f32).
// One wave per 16x16 output tile; 4 waves per 128-thread block share one
// column tile (nt), so the B (weight) panel is staged once per block in LDS,
// pre-converted to f16 and laid out in exact fragment order: each lane reads
// its whole B fragment as one aligned 32-byte LDS vector load.
// The im2col offset LUT is likewise permuted into fragment order in LDS
// (KP ints; all lanes of a half-group broadcast-read the same 64B line).
// K is padded to 64 with zero B columns, and A rows are clamped to M-1, so
// the steady-state loop has NO conditionals: clustered global_load_b32 +
// ds_load_b128 + cvt + two back-to-back v_wmma per 64-K step.
// Epilogue fuses bias + optional PReLU. f32 in memory, f16 into the matrix core.
// ---------------------------------------------------------------------------
__global__ __launch_bounds__(128)
void k_wmma_conv(const float* __restrict__ in, const float* __restrict__ wgt,
                 const float* __restrict__ bias, const float* __restrict__ alpha,
                 float* __restrict__ out,
                 int NIMG, int CIN, int IH, int IW, int COUT,
                 int KH, int KW, int OH, int OW)
{
    __shared__ _Float16 bpanel[MAX_KP * 16];   // [chunk][lane][i], chunk = k/32
    __shared__ int      offsP[MAX_KP];         // [chunk][half][i] fragment-ordered

    const int lane = threadIdx.x & 31;
    const int wid  = threadIdx.x >> 5;
    const int half = lane >> 4;                // 0: lanes 0-15, 1: lanes 16-31
    const int l16  = lane & 15;

    const int OHW  = OH * OW;
    const long M   = (long)NIMG * OHW;
    const int  K   = CIN * KH * KW;
    const int  KP  = (K + 63) & ~63;           // zero-padded K (multiple of 64)
    const int  KHW = KH * KW;
    const int  planeStride = IH * IW;
    const int  nt = blockIdx.y;

    // ---- fragment-ordered im2col LUT: offsP[chunk*32 + h*16 + i] ----------
    // A slot (h,i) -> k = chunk*32 + (i&8?16:0) + 8*h + 2*((i>>1)&3) + (i&1)
    for (int s = threadIdx.x; s < KP; s += 128) {
        int chunk = s >> 5, rem = s & 31;
        int h = rem >> 4, i = rem & 15;
        int kaRel = ((i & 8) ? 16 : 0) + h * 8 + (((i >> 1) & 3) << 1) + (i & 1);
        int k = chunk * 32 + kaRel;
        int o = 0;
        if (k < K) {
            int ci = k / KHW;
            int rr = k - ci * KHW;
            int kh = rr / KW;
            int kw = rr - kh * KW;
            o = ci * planeStride + kh * IW + kw;
        }
        offsP[s] = o;                          // k >= K -> offset 0 (valid addr,
    }                                          // product killed by zero B)

    // ---- B panel, f16, fragment order: bpanel[chunk*512 + lane*16 + i] ----
    // B slot (lane,i) -> col = nt*16 + (lane&15), k = chunk*32 + (i&8?16:0) + 8*(lane>>4) + (i&7)
    for (int e = threadIdx.x; e < KP * 16; e += 128) {
        int i = e & 15, ln = (e >> 4) & 31, chunk = e >> 9;
        int k   = chunk * 32 + ((i & 8) ? 16 : 0) + ((ln >> 4) << 3) + (i & 7);
        int col = nt * 16 + (ln & 15);
        float bv = (col < COUT && k < K) ? wgt[(long)col * K + k] : 0.f;
        bpanel[e] = (_Float16)bv;
    }
    __syncthreads();

    const long mt = (long)blockIdx.x * 4 + wid;
    if (mt * 16 >= M) return;                 // wave-uniform: EXEC all-ones for WMMA

    // A row, clamped so all loads are in-bounds (garbage rows discarded at store)
    long arow = mt * 16 + l16;
    if (arow >= M) arow = M - 1;
    const int img = (int)(arow / OHW);
    {
        // nothing
    }
    const int r0  = (int)(arow - (long)img * OHW);
    const int oy  = r0 / OW;
    const int ox  = r0 - oy * OW;
    const long inBase = (long)img * CIN * planeStride + (long)oy * IW + ox;

    const int  bcol = nt * 16 + l16;
    const bool bcol_ok = bcol < COUT;

    __builtin_prefetch(in + inBase, 0, 1);

    v8f acc = {};
    const int iters = KP >> 6;
    for (int it = 0; it < iters; ++it) {
        const int c0 = it * 2, c1 = c0 + 1;
        const int* op0 = offsP + c0 * 32 + half * 16;   // 64B-aligned, broadcast
        const int* op1 = offsP + c1 * 32 + half * 16;
        float av0[16], av1[16];
#pragma unroll
        for (int i = 0; i < 16; ++i) av0[i] = in[inBase + op0[i]];
#pragma unroll
        for (int i = 0; i < 16; ++i) av1[i] = in[inBase + op1[i]];
        v16h a0, a1;
#pragma unroll
        for (int i = 0; i < 16; ++i) {
            a0[i] = (_Float16)av0[i];
            a1[i] = (_Float16)av1[i];
        }
        const v16h b0 = *(const v16h*)(bpanel + c0 * 512 + lane * 16);
        const v16h b1 = *(const v16h*)(bpanel + c1 * 512 + lane * 16);
        acc = __builtin_amdgcn_wmma_f32_16x16x32_f16(false, a0, false, b0,
                                                     (short)0, acc, false, false);
        acc = __builtin_amdgcn_wmma_f32_16x16x32_f16(false, a1, false, b1,
                                                     (short)0, acc, false, false);
    }

    // C/D layout: VGPR r -> row r + 8*half, lane -> column
    if (bcol_ok) {
        const float bs = bias  ? bias[bcol]  : 0.f;
        const float al = alpha ? alpha[bcol] : 0.f;
#pragma unroll
        for (int r = 0; r < 8; ++r) {
            const long m = mt * 16 + r + half * 8;
            if (m < M) {
                float vv = acc[r] + bs;
                if (alpha) vv = vv > 0.f ? vv : al * vv;
                int im = (int)(m / OHW);
                int rr = (int)(m - (long)im * OHW);
                out[((long)im * COUT + bcol) * OHW + rr] = vv;   // NCHW
            }
        }
    }
}

// ---------------------------------------------------------------------------
// Max-pool (reduce_window with -inf padding), NCHW
// ---------------------------------------------------------------------------
__global__ void k_maxpool(const float* __restrict__ in, float* __restrict__ out,
                          int NC, int IH, int IW, int OH, int OW, int k, int s, int p)
{
    long t = (long)blockIdx.x * blockDim.x + threadIdx.x;
    long total = (long)NC * OH * OW;
    if (t >= total) return;
    int ox = (int)(t % OW); long r = t / OW;
    int oy = (int)(r % OH); int nc = (int)(r / OH);
    float m = -__builtin_inff();
    int y0 = oy * s - p, x0 = ox * s - p;
    for (int dy = 0; dy < k; ++dy) {
        int y = y0 + dy; if (y < 0 || y >= IH) continue;
        for (int dx = 0; dx < k; ++dx) {
            int x = x0 + dx; if (x < 0 || x >= IW) continue;
            float v = in[((long)nc * IH + y) * IW + x];
            m = v > m ? v : m;
        }
    }
    out[t] = m;
}

// ---------------------------------------------------------------------------
// 2-way softmax score: score[b*HW+i] = softmax(cls[b,:,i])[0]  (cls NCHW, C=2)
// ---------------------------------------------------------------------------
__global__ void k_softmax2_score(const float* __restrict__ cls, float* __restrict__ score,
                                 int HW, long total)
{
    long t = (long)blockIdx.x * blockDim.x + threadIdx.x;
    if (t >= total) return;
    int b = (int)(t / HW), i = (int)(t % HW);
    float c0 = cls[((long)b * 2 + 0) * HW + i];
    float c1 = cls[((long)b * 2 + 1) * HW + i];
    float m = fmaxf(c0, c1);
    float e0 = expf(c0 - m), e1 = expf(c1 - m);
    score[t] = e0 / (e0 + e1);
}

// ---------------------------------------------------------------------------
// TopK per batch row: iterative argmax with lowest-index tie break (lax.top_k)
// ---------------------------------------------------------------------------
__global__ void k_topk(const float* __restrict__ score, int* __restrict__ taken,
                       int* __restrict__ outIdx, int Ncand, int Ksel)
{
    const int b = blockIdx.x;
    const float* s = score + (long)b * Ncand;
    int* tk = taken + (long)b * Ncand;
    __shared__ float sv[256];
    __shared__ int   si[256];
    for (int i = threadIdx.x; i < Ncand; i += blockDim.x) tk[i] = 0;
    __syncthreads();
    for (int k = 0; k < Ksel; ++k) {
        float bestv = -__builtin_inff(); int besti = 0x7fffffff;
        for (int i = threadIdx.x; i < Ncand; i += blockDim.x) {
            if (!tk[i]) {
                float v = s[i];
                if (v > bestv || (v == bestv && i < besti)) { bestv = v; besti = i; }
            }
        }
        sv[threadIdx.x] = bestv; si[threadIdx.x] = besti;
        __syncthreads();
        for (int offd = 128; offd > 0; offd >>= 1) {
            if (threadIdx.x < offd) {
                float v2 = sv[threadIdx.x + offd]; int i2 = si[threadIdx.x + offd];
                if (v2 > sv[threadIdx.x] ||
                    (v2 == sv[threadIdx.x] && i2 < si[threadIdx.x])) {
                    sv[threadIdx.x] = v2; si[threadIdx.x] = i2;
                }
            }
            __syncthreads();
        }
        if (threadIdx.x == 0) { outIdx[b * Ksel + k] = si[0]; tk[si[0]] = 1; }
        __syncthreads();
    }
}

// ---------------------------------------------------------------------------
// Stage-1 boxes: gather bb/lm at topk indices, translate (start=2*pos, scale=12),
// then b1 += l1 @ bbr_w.T + bbr_b
// ---------------------------------------------------------------------------
__global__ void k_stage1(const int* __restrict__ idx1,
                         const float* __restrict__ bbh, const float* __restrict__ lmh,
                         const float* __restrict__ bbrW, const float* __restrict__ bbrB,
                         float* __restrict__ b1buf)
{
    int t = blockIdx.x * blockDim.x + threadIdx.x;
    if (t >= 8 * 512) return;
    int b = t >> 9;
    int i = idx1[t];
    float sx = 2.f * (float)(i % 251);
    float sy = 2.f * (float)(i / 251);
    float bb[4], lm[10];
    for (int c = 0; c < 4; ++c)  bb[c] = bbh[((long)b * 4 + c) * 63001 + i] * 12.f;
    bb[0] += sx; bb[1] += sy;
    for (int j = 0; j < 10; ++j)
        lm[j] = lmh[((long)b * 10 + j) * 63001 + i] * 12.f + ((j & 1) ? sy : sx);
    for (int c = 0; c < 4; ++c) {
        float v = bb[c] + bbrB[c];
        for (int j = 0; j < 10; ++j) v += lm[j] * bbrW[c * 10 + j];
        b1buf[t * 4 + c] = v;
    }
}

// ---------------------------------------------------------------------------
// Bilinear crop_resize from x (B,3,512,512); boxes (NB,4) = (x0,y0,bw,bh)
// ---------------------------------------------------------------------------
__global__ void k_crop(const float* __restrict__ img, const float* __restrict__ boxes,
                       float* __restrict__ out, int NB, int PB, int S)
{
    long t = (long)blockIdx.x * blockDim.x + threadIdx.x;
    long total = (long)NB * 3 * S * S;
    if (t >= total) return;
    int xo = (int)(t % S); long r = t / S;
    int yo = (int)(r % S); r /= S;
    int c  = (int)(r % 3);
    int n  = (int)(r / 3);
    int ib = n / PB;
    float x0 = boxes[n * 4 + 0], y0 = boxes[n * 4 + 1];
    float bw = boxes[n * 4 + 2], bh = boxes[n * 4 + 3];
    float gx = fminf(fmaxf(x0 + (xo + 0.5f) / (float)S * bw, 0.f), 511.f);
    float gy = fminf(fmaxf(y0 + (yo + 0.5f) / (float)S * bh, 0.f), 511.f);
    int xi = (int)floorf(gx); int xj = min(xi + 1, 511);
    int yi = (int)floorf(gy); int yj = min(yi + 1, 511);
    float tx = gx - (float)xi, ty = gy - (float)yi;
    const float* ip = img + ((long)ib * 3 + c) * 512 * 512;
    float v00 = ip[yi * 512 + xi], v01 = ip[yi * 512 + xj];
    float v10 = ip[yj * 512 + xi], v11 = ip[yj * 512 + xj];
    out[t] = (1.f - ty) * ((1.f - tx) * v00 + tx * v01) +
             ty        * ((1.f - tx) * v10 + tx * v11);
}

// ---------------------------------------------------------------------------
// RNet collect: softmax score + translate with (start=b1[:2], scale=b1[2:4]/24)
// heads laid out (N, C) row-major (dense output, OH=OW=1)
// ---------------------------------------------------------------------------
__global__ void k_rnet_collect(const float* __restrict__ cl, const float* __restrict__ bbh,
                               const float* __restrict__ lmh, const float* __restrict__ b1buf,
                               float* __restrict__ score2, float* __restrict__ b2full,
                               float* __restrict__ l2full)
{
    int n = blockIdx.x * blockDim.x + threadIdx.x;
    if (n >= 4096) return;
    float c0 = cl[n * 2 + 0], c1 = cl[n * 2 + 1];
    float m = fmaxf(c0, c1);
    float e0 = expf(c0 - m), e1 = expf(c1 - m);
    score2[n] = e0 / (e0 + e1);
    float sx = b1buf[n * 4 + 0], sy = b1buf[n * 4 + 1];
    float sw = b1buf[n * 4 + 2] / 24.f, sh = b1buf[n * 4 + 3] / 24.f;
    b2full[n * 4 + 0] = bbh[n * 4 + 0] * sw + sx;
    b2full[n * 4 + 1] = bbh[n * 4 + 1] * sh + sy;
    b2full[n * 4 + 2] = bbh[n * 4 + 2] * sw;
    b2full[n * 4 + 3] = bbh[n * 4 + 3] * sh;
    for (int j = 0; j < 10; ++j)
        l2full[n * 10 + j] = lmh[n * 10 + j] * ((j & 1) ? sh : sw) + ((j & 1) ? sy : sx);
}

__global__ void k_stage2_gather(const int* __restrict__ idx2,
                                const float* __restrict__ b2full, const float* __restrict__ l2full,
                                const float* __restrict__ bbrW, const float* __restrict__ bbrB,
                                float* __restrict__ b2buf)
{
    int t = blockIdx.x * blockDim.x + threadIdx.x;
    if (t >= 8 * 128) return;
    int b = t >> 7;
    int src = (b << 9) + idx2[t];
    for (int c = 0; c < 4; ++c) {
        float v = b2full[src * 4 + c] + bbrB[c];
        for (int j = 0; j < 10; ++j) v += l2full[src * 10 + j] * bbrW[c * 10 + j];
        b2buf[t * 4 + c] = v;
    }
}

__global__ void k_onet_collect(const float* __restrict__ cl, const float* __restrict__ bbh,
                               const float* __restrict__ lmh, const float* __restrict__ b2buf,
                               float* __restrict__ score3, float* __restrict__ c3full,
                               float* __restrict__ b3full, float* __restrict__ l3full)
{
    int n = blockIdx.x * blockDim.x + threadIdx.x;
    if (n >= 1024) return;
    float c0 = cl[n * 2 + 0], c1 = cl[n * 2 + 1];
    float m = fmaxf(c0, c1);
    float e0 = expf(c0 - m), e1 = expf(c1 - m);
    float s0 = e0 / (e0 + e1);
    score3[n] = s0;
    c3full[n * 2 + 0] = s0;
    c3full[n * 2 + 1] = e1 / (e0 + e1);
    float sx = b2buf[n * 4 + 0], sy = b2buf[n * 4 + 1];
    float sw = b2buf[n * 4 + 2] / 48.f, sh = b2buf[n * 4 + 3] / 48.f;
    b3full[n * 4 + 0] = bbh[n * 4 + 0] * sw + sx;
    b3full[n * 4 + 1] = bbh[n * 4 + 1] * sh + sy;
    b3full[n * 4 + 2] = bbh[n * 4 + 2] * sw;
    b3full[n * 4 + 3] = bbh[n * 4 + 3] * sh;
    for (int j = 0; j < 10; ++j)
        l3full[n * 10 + j] = lmh[n * 10 + j] * ((j & 1) ? sh : sw) + ((j & 1) ? sy : sx);
}

__global__ void k_stage3_out(const int* __restrict__ idx3,
                             const float* __restrict__ c3full, const float* __restrict__ b3full,
                             const float* __restrict__ l3full,
                             const float* __restrict__ bbrW, const float* __restrict__ bbrB,
                             float* __restrict__ out)
{
    int t = blockIdx.x * blockDim.x + threadIdx.x;
    if (t >= 8 * 64) return;
    int b = t >> 6;
    int src = (b << 7) + idx3[t];
    out[t * 2 + 0] = c3full[src * 2 + 0];                 // c3: (8,64,2) at offset 0
    out[t * 2 + 1] = c3full[src * 2 + 1];
    for (int c = 0; c < 4; ++c) {                         // b3: (8,64,4) at 1024
        float v = b3full[src * 4 + c] + bbrB[c];
        for (int j = 0; j < 10; ++j) v += l3full[src * 10 + j] * bbrW[c * 10 + j];
        out[1024 + t * 4 + c] = v;
    }
    for (int j = 0; j < 10; ++j)                          // l3: (8,64,10) at 3072
        out[3072 + t * 10 + j] = l3full[src * 10 + j];
}

// ---------------------------------------------------------------------------
// Host-side launchers
// ---------------------------------------------------------------------------
static void launch_conv(const float* in, const float* w, const float* b, const float* a,
                        float* out, int NIMG, int CIN, int IH, int IW, int COUT,
                        int KH, int KW, hipStream_t s)
{
    int OH = IH - KH + 1, OW = IW - KW + 1;
    long M = (long)NIMG * OH * OW;
    long mt = (M + 15) / 16;
    int  nt = (COUT + 15) / 16;
    dim3 grid((unsigned)((mt + 3) / 4), (unsigned)nt, 1);
    k_wmma_conv<<<grid, dim3(128, 1, 1), 0, s>>>(in, w, b, a, out,
                                                 NIMG, CIN, IH, IW, COUT, KH, KW, OH, OW);
}

static void launch_pool(const float* in, float* out, int NC, int IH, int IW,
                        int k, int st, int p, hipStream_t s)
{
    int OH = (IH + 2 * p - k) / st + 1, OW = (IW + 2 * p - k) / st + 1;
    long total = (long)NC * OH * OW;
    k_maxpool<<<(unsigned)((total + 255) / 256), 256, 0, s>>>(in, out, NC, IH, IW, OH, OW, k, st, p);
}

extern "C" void kernel_launch(void* const* d_in, const int* in_sizes, int n_in,
                              void* d_out, int out_size, void* d_ws, size_t ws_size,
                              hipStream_t stream)
{
    (void)in_sizes; (void)n_in; (void)out_size; (void)ws_size;
    const float* x = (const float*)d_in[0];
    // PNet params
    const float *p_c1w = (const float*)d_in[1],  *p_c1b = (const float*)d_in[2],  *p_a1 = (const float*)d_in[3];
    const float *p_c2w = (const float*)d_in[4],  *p_c2b = (const float*)d_in[5],  *p_a2 = (const float*)d_in[6];
    const float *p_c3w = (const float*)d_in[7],  *p_c3b = (const float*)d_in[8],  *p_a3 = (const float*)d_in[9];
    const float *p_clw = (const float*)d_in[10], *p_clb = (const float*)d_in[11];
    const float *p_bbw = (const float*)d_in[12], *p_bbb = (const float*)d_in[13];
    const float *p_lmw = (const float*)d_in[14], *p_lmb = (const float*)d_in[15];
    // RNet params
    const float *r_c1w = (const float*)d_in[16], *r_c1b = (const float*)d_in[17], *r_a1 = (const float*)d_in[18];
    const float *r_c2w = (const float*)d_in[19], *r_c2b = (const float*)d_in[20], *r_a2 = (const float*)d_in[21];
    const float *r_c3w = (const float*)d_in[22], *r_c3b = (const float*)d_in[23], *r_a3 = (const float*)d_in[24];
    const float *r_fw  = (const float*)d_in[25], *r_fb  = (const float*)d_in[26], *r_a4 = (const float*)d_in[27];
    const float *r_clw = (const float*)d_in[28], *r_clb = (const float*)d_in[29];
    const float *r_bbw = (const float*)d_in[30], *r_bbb = (const float*)d_in[31];
    const float *r_lmw = (const float*)d_in[32], *r_lmb = (const float*)d_in[33];
    // ONet params
    const float *o_c1w = (const float*)d_in[34], *o_c1b = (const float*)d_in[35], *o_a1 = (const float*)d_in[36];
    const float *o_c2w = (const float*)d_in[37], *o_c2b = (const float*)d_in[38], *o_a2 = (const float*)d_in[39];
    const float *o_c3w = (const float*)d_in[40], *o_c3b = (const float*)d_in[41], *o_a3 = (const float*)d_in[42];
    const float *o_c4w = (const float*)d_in[43], *o_c4b = (const float*)d_in[44], *o_a4 = (const float*)d_in[45];
    const float *o_fw  = (const float*)d_in[46], *o_fb  = (const float*)d_in[47], *o_a5 = (const float*)d_in[48];
    const float *o_clw = (const float*)d_in[49], *o_clb = (const float*)d_in[50];
    const float *o_bbw = (const float*)d_in[51], *o_bbb = (const float*)d_in[52];
    const float *o_lmw = (const float*)d_in[53], *o_lmb = (const float*)d_in[54];
    const float *bbrW  = (const float*)d_in[55], *bbrB  = (const float*)d_in[56];
    float* out = (float*)d_out;

    // ---- workspace arena (256B-aligned bump allocator) ----
    char* ws = (char*)d_ws;
    size_t off = 0;
    auto alloc = [&](size_t nElems) -> float* {
        float* p = (float*)(ws + off);
        off += ((nElems * 4 + 255) & ~(size_t)255);
        return p;
    };
    float* score1 = alloc((size_t)8 * 63001);
    int*   taken  = (int*)alloc((size_t)8 * 63001);
    int*   idx1   = (int*)alloc(8 * 512);
    float* b1buf  = alloc(8 * 512 * 4);
    float* score2 = alloc(4096);
    float* b2full = alloc(4096 * 4);
    float* l2full = alloc(4096 * 10);
    int*   idx2   = (int*)alloc(8 * 128);
    float* b2buf  = alloc(1024 * 4);
    float* score3 = alloc(1024);
    float* c3full = alloc(1024 * 2);
    float* b3full = alloc(1024 * 4);
    float* l3full = alloc(1024 * 10);
    int*   idx3   = (int*)alloc(8 * 64);
    float* slotC  = alloc(8500000);     // heads (pnet: cls/bb/lm at 251x251)
    float* slotA  = alloc(21000000);    // ping
    float* slotB  = alloc(70000000);    // pong (ONet conv1 = 69.34M elems)
    float* cls = slotC;
    float* bbh = slotC + 1100000;
    float* lmh = slotC + 3200000;

    // =================== PNet (8,3,512,512) ===================
    launch_conv(x,     p_c1w, p_c1b, p_a1, slotA, 8, 3,  512, 512, 10, 3, 3, stream); // ->510
    launch_pool(slotA, slotB, 8 * 10, 510, 510, 2, 2, 0, stream);                     // ->255
    launch_conv(slotB, p_c2w, p_c2b, p_a2, slotA, 8, 10, 255, 255, 16, 3, 3, stream); // ->253
    launch_conv(slotA, p_c3w, p_c3b, p_a3, slotB, 8, 16, 253, 253, 32, 3, 3, stream); // ->251
    launch_conv(slotB, p_clw, p_clb, nullptr, cls, 8, 32, 251, 251, 2,  1, 1, stream);
    launch_conv(slotB, p_bbw, p_bbb, nullptr, bbh, 8, 32, 251, 251, 4,  1, 1, stream);
    launch_conv(slotB, p_lmw, p_lmb, nullptr, lmh, 8, 32, 251, 251, 10, 1, 1, stream);
    k_softmax2_score<<<(unsigned)(((long)8 * 63001 + 255) / 256), 256, 0, stream>>>(cls, score1, 63001, (long)8 * 63001);
    k_topk<<<8, 256, 0, stream>>>(score1, taken, idx1, 63001, 512);
    k_stage1<<<16, 256, 0, stream>>>(idx1, bbh, lmh, bbrW, bbrB, b1buf);
    k_crop<<<(unsigned)(((long)4096 * 3 * 576 + 255) / 256), 256, 0, stream>>>(x, b1buf, slotA, 4096, 512, 24);

    // =================== RNet (4096,3,24,24) ===================
    launch_conv(slotA, r_c1w, r_c1b, r_a1, slotB, 4096, 3,   24, 24, 28,  3, 3, stream); // ->22
    launch_pool(slotB, slotA, 4096 * 28, 22, 22, 2, 2, 0, stream);                       // ->11
    launch_conv(slotA, r_c2w, r_c2b, r_a2, slotB, 4096, 28,  11, 11, 48,  3, 3, stream); // ->9
    launch_pool(slotB, slotA, 4096 * 48, 9, 9, 3, 2, 0, stream);                         // ->4
    launch_conv(slotA, r_c3w, r_c3b, r_a3, slotB, 4096, 48,  4,  4,  64,  2, 2, stream); // ->3
    launch_conv(slotB, r_fw,  r_fb,  r_a4, slotA, 4096, 576, 1,  1,  128, 1, 1, stream); // fc
    launch_conv(slotA, r_clw, r_clb, nullptr, cls, 4096, 128, 1, 1, 2,  1, 1, stream);
    launch_conv(slotA, r_bbw, r_bbb, nullptr, bbh, 4096, 128, 1, 1, 4,  1, 1, stream);
    launch_conv(slotA, r_lmw, r_lmb, nullptr, lmh, 4096, 128, 1, 1, 10, 1, 1, stream);
    k_rnet_collect<<<16, 256, 0, stream>>>(cls, bbh, lmh, b1buf, score2, b2full, l2full);
    k_topk<<<8, 256, 0, stream>>>(score2, taken, idx2, 512, 128);
    k_stage2_gather<<<4, 256, 0, stream>>>(idx2, b2full, l2full, bbrW, bbrB, b2buf);
    k_crop<<<(unsigned)(((long)1024 * 3 * 2304 + 255) / 256), 256, 0, stream>>>(x, b2buf, slotA, 1024, 128, 48);

    // =================== ONet (1024,3,48,48) ===================
    launch_conv(slotA, o_c1w, o_c1b, o_a1, slotB, 1024, 3,    48, 48, 32,  3, 3, stream); // ->46
    launch_pool(slotB, slotA, 1024 * 32, 46, 46, 3, 2, 1, stream);                        // ->23
    launch_conv(slotA, o_c2w, o_c2b, o_a2, slotB, 1024, 32,   23, 23, 64,  3, 3, stream); // ->21
    launch_pool(slotB, slotA, 1024 * 64, 21, 21, 3, 2, 0, stream);                        // ->10
    launch_conv(slotA, o_c3w, o_c3b, o_a3, slotB, 1024, 64,   10, 10, 64,  3, 3, stream); // ->8
    launch_pool(slotB, slotA, 1024 * 64, 8, 8, 2, 2, 0, stream);                          // ->4
    launch_conv(slotA, o_c4w, o_c4b, o_a4, slotB, 1024, 64,   4,  4,  128, 2, 2, stream); // ->3
    launch_conv(slotB, o_fw,  o_fb,  o_a5, slotA, 1024, 1152, 1,  1,  256, 1, 1, stream); // fc
    launch_conv(slotA, o_clw, o_clb, nullptr, cls, 1024, 256, 1, 1, 2,  1, 1, stream);
    launch_conv(slotA, o_bbw, o_bbb, nullptr, bbh, 1024, 256, 1, 1, 4,  1, 1, stream);
    launch_conv(slotA, o_lmw, o_lmb, nullptr, lmh, 1024, 256, 1, 1, 10, 1, 1, stream);
    k_onet_collect<<<4, 256, 0, stream>>>(cls, bbh, lmh, b2buf, score3, c3full, b3full, l3full);
    k_topk<<<8, 256, 0, stream>>>(score3, taken, idx3, 128, 64);
    k_stage3_out<<<2, 256, 0, stream>>>(idx3, c3full, b3full, l3full, bbrW, bbrB, out);
}
